// ModularUnit_66554813218910
// MI455X (gfx1250) — matrix operand order repaired and checked by reading, not compiled, not run on hardware
//
#include <hip/hip_runtime.h>

// LIF recurrent network, 50 steps, 8192 neurons.
// Strategy (MI455X): bf16 weight copies (w_rec bf16 = 128MB fits 192MB L2),
// one up-front WMMA GEMM for all feedforward currents, then 50 sequential
// WMMA mat-vecs (z broadcast into A rows) + fused LIF state update.
// Workspace requirement: ~259 MB.

#define NN 8192
#define TSTEPS 50

#define DT_TAU_MEM 0.00005f   // DT * TAU_MEM_INV = 0.001/20
#define DT_TAU_SYN 0.0001f    // DT * TAU_SYN_INV = 0.001/10
#define V_LEAK_C  (-70.0f)
#define V_TH_C    (-55.0f)
#define V_RESET_C (-70.0f)

typedef __attribute__((ext_vector_type(16))) __bf16 v16bf;
typedef __attribute__((ext_vector_type(8)))  __bf16 v8bf;
typedef __attribute__((ext_vector_type(8)))  float  v8f;
typedef __attribute__((ext_vector_type(4)))  float  vf4;

__device__ __forceinline__ unsigned short f2bf(float f) {
    unsigned int u = __float_as_uint(f);
    unsigned int r = u + 0x7FFFu + ((u >> 16) & 1u);   // round-to-nearest-even
    return (unsigned short)(r >> 16);
}

// ---------------------------------------------------------------- init state
__global__ void lif_init(float* __restrict__ v, float* __restrict__ i_cur,
                         float* __restrict__ rec, unsigned short* __restrict__ z_bf) {
    int n = blockIdx.x * 256 + threadIdx.x;
    if (n < NN) {
        v[n] = V_LEAK_C;
        i_cur[n] = 0.0f;
        rec[n] = 0.0f;
        z_bf[n] = 0;
    }
}

// ------------------------------------------------- fp32 -> bf16 weight copies
// grid.y selects matrix; nontemporal fp32 reads keep L2 free for bf16 copies.
__global__ __launch_bounds__(256) void convert_w(const float* __restrict__ w_in,
                                                 const float* __restrict__ w_rec,
                                                 unsigned short* __restrict__ win_bf,
                                                 unsigned short* __restrict__ wrec_bf) {
    const float* src = blockIdx.y ? w_rec : w_in;
    unsigned short* dst = blockIdx.y ? wrec_bf : win_bf;
    size_t base = ((size_t)blockIdx.x * 256 + threadIdx.x) * 8;
    vf4 a0 = __builtin_nontemporal_load(reinterpret_cast<const vf4*>(src + base));
    vf4 a1 = __builtin_nontemporal_load(reinterpret_cast<const vf4*>(src + base + 4));
    unsigned short r[8];
    r[0] = f2bf(a0.x); r[1] = f2bf(a0.y); r[2] = f2bf(a0.z); r[3] = f2bf(a0.w);
    r[4] = f2bf(a1.x); r[5] = f2bf(a1.y); r[6] = f2bf(a1.z); r[7] = f2bf(a1.w);
    uint4 packed;
    packed.x = (unsigned)r[0] | ((unsigned)r[1] << 16);
    packed.y = (unsigned)r[2] | ((unsigned)r[3] << 16);
    packed.z = (unsigned)r[4] | ((unsigned)r[5] << 16);
    packed.w = (unsigned)r[6] | ((unsigned)r[7] << 16);
    *reinterpret_cast<uint4*>(dst + base) = packed;
}

__global__ void convert_x(const float* __restrict__ x, unsigned short* __restrict__ x_bf) {
    int idx = blockIdx.x * 256 + threadIdx.x;
    if (idx < TSTEPS * NN) x_bf[idx] = f2bf(x[idx]);
}

// ------------------------------------------------------------ feedforward GEMM
// FF[t][n] = sum_k x[t][k] * w_in[n][k]. M padded to 64 (4 tiles of 16).
// Block: 8 waves = 4 N-subtiles x 2 M-subtiles. Full K per wave (no atomics).
__global__ __launch_bounds__(256) void ff_gemm(const unsigned short* __restrict__ win_bf,
                                               const unsigned short* __restrict__ x_bf,
                                               float* __restrict__ FF) {
    const int tid  = threadIdx.x;
    const int wave = tid >> 5, lane = tid & 31;
    const int nsub = wave & 3, msub = wave >> 2;
    const int nb = blockIdx.x * 64 + nsub * 16;
    const int t0 = blockIdx.y * 32 + msub * 16;
    const int g  = (lane & 16) ? 16 : 0;      // A-operand K sub-offset per half-wave
    const int nl = lane & 15;

    const unsigned short* wrow = win_bf + (size_t)(nb + nl) * NN + ((lane & 16) ? 16 : 0);
    const __bf16* xb = reinterpret_cast<const __bf16*>(x_bf + (size_t)(t0 + nl) * NN);

    v8f c0 = {}, c1 = {};
#pragma unroll 4
    for (int kk = 0; kk < NN; kk += 64) {
        // ---- k-step 0
        v8bf alo = *reinterpret_cast<const v8bf*>(xb + kk + g);
        v8bf ahi = *reinterpret_cast<const v8bf*>(xb + kk + 8 + g);
        v16bf a = __builtin_shufflevector(alo, ahi, 0,1,2,3,4,5,6,7,8,9,10,11,12,13,14,15);
        v16bf b = *reinterpret_cast<const v16bf*>(wrow + kk);
        c0 = __builtin_amdgcn_wmma_f32_16x16x32_bf16(false, a, false, b, (short)0, c0, false, false);
        // ---- k-step 1 (second accumulator for ILP)
        v8bf alo2 = *reinterpret_cast<const v8bf*>(xb + kk + 32 + g);
        v8bf ahi2 = *reinterpret_cast<const v8bf*>(xb + kk + 40 + g);
        v16bf a2 = __builtin_shufflevector(alo2, ahi2, 0,1,2,3,4,5,6,7,8,9,10,11,12,13,14,15);
        v16bf b2 = *reinterpret_cast<const v16bf*>(wrow + kk + 32);
        c1 = __builtin_amdgcn_wmma_f32_16x16x32_bf16(false, a2, false, b2, (short)0, c1, false, false);
    }
    v8f c = c0 + c1;
#pragma unroll
    for (int r = 0; r < 8; ++r) {
        int t = t0 + ((lane & 16) ? 8 + r : r);
        if (t < TSTEPS) FF[(size_t)t * NN + nb + nl] = c[r];
    }
}

// ---------------------------------------------------------- recurrent mat-vec
// rec[n] += sum_k z[k] * w_rec[n][k].  z broadcast into all 16 A rows; one
// wmma ingests a 32x16 weight tile. K split over blockIdx.y -> f32 atomics.
#define KSPLIT 4
#define KCHUNK (NN / KSPLIT)   // 2048
__global__ __launch_bounds__(256) void rec_matvec(const unsigned short* __restrict__ wrec_bf,
                                                  const unsigned short* __restrict__ z_bf,
                                                  float* __restrict__ rec) {
    __shared__ unsigned short zs[KCHUNK];
    const int tid = threadIdx.x;
    const int k0  = blockIdx.y * KCHUNK;
    // Stage the z chunk in LDS: 4 KB, one b128 per thread.
    reinterpret_cast<uint4*>(zs)[tid] = reinterpret_cast<const uint4*>(z_bf + k0)[tid];
    __syncthreads();

    const int wave = tid >> 5, lane = tid & 31;
    const int nb = blockIdx.x * 128 + wave * 16;
    const int g  = (lane & 16) ? 16 : 0;
    const int nl = lane & 15;
    const unsigned short* wrow = wrec_bf + (size_t)(nb + nl) * NN + k0 + ((lane & 16) ? 16 : 0);
    const __bf16* zb = reinterpret_cast<const __bf16*>(zs);

    v8f c0 = {}, c1 = {};
#pragma unroll 4
    for (int kk = 0; kk < KCHUNK; kk += 64) {
        v8bf alo = *reinterpret_cast<const v8bf*>(zb + kk + g);
        v8bf ahi = *reinterpret_cast<const v8bf*>(zb + kk + 8 + g);
        v16bf a = __builtin_shufflevector(alo, ahi, 0,1,2,3,4,5,6,7,8,9,10,11,12,13,14,15);
        v16bf b = *reinterpret_cast<const v16bf*>(wrow + kk);
        c0 = __builtin_amdgcn_wmma_f32_16x16x32_bf16(false, a, false, b, (short)0, c0, false, false);

        v8bf alo2 = *reinterpret_cast<const v8bf*>(zb + kk + 32 + g);
        v8bf ahi2 = *reinterpret_cast<const v8bf*>(zb + kk + 40 + g);
        v16bf a2 = __builtin_shufflevector(alo2, ahi2, 0,1,2,3,4,5,6,7,8,9,10,11,12,13,14,15);
        v16bf b2 = *reinterpret_cast<const v16bf*>(wrow + kk + 32);
        c1 = __builtin_amdgcn_wmma_f32_16x16x32_bf16(false, a2, false, b2, (short)0, c1, false, false);
    }
    // All A rows equal -> all D rows equal; row 0 lives in c[0], lanes 0..15.
    float s = c0[0] + c1[0];
    if (lane < 16) atomicAdd(&rec[nb + lane], s);
}

// --------------------------------------------------------------- state update
// Consumes rec (and re-zeroes it for the next step), emits z (fp32 out + bf16).
__global__ void lif_state(const float* __restrict__ FFt, float* __restrict__ rec,
                          float* __restrict__ v, float* __restrict__ i_cur,
                          unsigned short* __restrict__ z_bf, float* __restrict__ out_t) {
    int n = blockIdx.x * 256 + threadIdx.x;
    if (n >= NN) return;
    float vv = v[n], ii = i_cur[n];
    float v_d = vv + DT_TAU_MEM * ((V_LEAK_C - vv) + ii);
    float i_d = ii - DT_TAU_SYN * ii;
    float z   = (v_d - V_TH_C > 0.0f) ? 1.0f : 0.0f;
    v[n] = (z > 0.0f) ? V_RESET_C : v_d;
    i_cur[n] = i_d + FFt[n] + rec[n];   // recurrent term used OLD z (computed before this kernel)
    rec[n] = 0.0f;                      // ready for next step's atomics
    out_t[n] = z;
    z_bf[n] = (z > 0.0f) ? (unsigned short)0x3F80 : (unsigned short)0;
}

// ------------------------------------------------------------------- launcher
extern "C" void kernel_launch(void* const* d_in, const int* in_sizes, int n_in,
                              void* d_out, int out_size, void* d_ws, size_t ws_size,
                              hipStream_t stream) {
    const float* spikes_in = (const float*)d_in[0];   // [50, 8192]
    const float* w_in      = (const float*)d_in[1];   // [8192, 8192]
    const float* w_rec     = (const float*)d_in[2];   // [8192, 8192]
    float* out = (float*)d_out;                       // [50, 8192]

    char* ws = (char*)d_ws;
    const size_t WBYTES = (size_t)NN * NN * 2;        // 128 MB each
    unsigned short* wrec_bf = (unsigned short*)(ws);
    unsigned short* win_bf  = (unsigned short*)(ws + WBYTES);
    unsigned short* x_bf    = (unsigned short*)(ws + 2 * WBYTES);                 // 64 rows alloc'd
    float*          FF      = (float*)(ws + 2 * WBYTES + (size_t)64 * NN * 2);    // 2 MB pad
    float*          rec     = (float*)(ws + 2 * WBYTES + (size_t)64 * NN * 2 + 2097152);
    unsigned short* z_bf    = (unsigned short*)((char*)rec + 32768);
    float*          v_st    = (float*)((char*)z_bf + 32768);
    float*          i_st    = (float*)((char*)v_st + 32768);

    // 1) state init
    lif_init<<<NN / 256, 256, 0, stream>>>(v_st, i_st, rec, z_bf);

    // 2) weight + input conversion to bf16
    convert_w<<<dim3((NN * (size_t)NN) / (256 * 8), 2), 256, 0, stream>>>(w_in, w_rec, win_bf, wrec_bf);
    convert_x<<<(TSTEPS * NN + 255) / 256, 256, 0, stream>>>(spikes_in, x_bf);

    // 3) all feedforward currents in one GEMM: FF = X @ w_in.T
    ff_gemm<<<dim3(NN / 64, 2), 256, 0, stream>>>(win_bf, x_bf, FF);

    // 4) 50 sequential steps: recurrent mat-vec (reads prev z) then LIF update
    for (int t = 0; t < TSTEPS; ++t) {
        rec_matvec<<<dim3(NN / 128, KSPLIT), 256, 0, stream>>>(wrec_bf, z_bf, rec);
        lif_state<<<NN / 256, 256, 0, stream>>>(FF + (size_t)t * NN, rec, v_st, i_st,
                                                z_bf, out + (size_t)t * NN);
    }
}
